// STRGCNLayer_3496103379384
// MI455X (gfx1250) — matrix operand order repaired and checked by reading, not compiled, not run on hardware
//
#include <hip/hip_runtime.h>
#include <hip/hip_bf16.h>
#include <math.h>

#define B_    8
#define L_    2048
#define CIN   64
#define COUT  64
#define DEMB  32
#define NNODE 2048

typedef __attribute__((ext_vector_type(16))) _Float16 v16h;
typedef __attribute__((ext_vector_type(8)))  _Float16 v8h;
typedef __attribute__((ext_vector_type(8)))  float    v8f;

// ---------------- workspace layout (bytes, all 1KB-aligned) ----------------
constexpr size_t OFF_EMBSUM = 0;                                   // B*32*4   = 1 KB
constexpr size_t OFF_DENOM  = OFF_EMBSUM + 1024;                   // B*L*4    = 64 KB
constexpr size_t OFF_SCALE  = OFF_DENOM  + (size_t)B_*L_*4;
constexpr size_t OFF_S0     = OFF_SCALE  + (size_t)B_*L_*4;
constexpr size_t OFF_S1     = OFF_S0     + (size_t)B_*L_*4;
constexpr size_t OFF_S2     = OFF_S1     + (size_t)B_*L_*4;
constexpr size_t OFF_TEMBH  = OFF_S2     + (size_t)B_*L_*4;        // f16 t_emb, 1 MB
constexpr size_t OFF_XT     = OFF_TEMBH  + (size_t)B_*L_*DEMB*2;   // f16 x^T (B,64,L), 2 MB
constexpr size_t OFF_WT     = OFF_XT     + (size_t)B_*CIN*L_*2;    // f32 weight_T, 4 MB
constexpr size_t OFF_T1T    = OFF_WT     + (size_t)B_*L_*COUT*4;   // f16 T1^T, 2 MB
constexpr size_t OFF_ADJ    = OFF_T1T    + (size_t)B_*COUT*L_*2;   // f16 adj_raw, 67 MB

// ------------------------------ k_sum: per-batch sum of t_emb rows ---------
__global__ void k_sum(const float* __restrict__ temb, float* __restrict__ embsum) {
    int b = blockIdx.x;
    int lane = threadIdx.x & 31;          // dim 0..31
    int wave = threadIdx.x >> 5;          // 0..7
    const float* base = temb + (size_t)b * L_ * DEMB;
    float acc = 0.f;
    for (int l = wave; l < L_; l += 8) acc += base[(size_t)l * DEMB + lane];
    __shared__ float red[8][32];
    red[wave][lane] = acc;
    __syncthreads();
    if (wave == 0) {
        float s = red[0][lane];
#pragma unroll
        for (int w = 1; w < 8; w++) s += red[w][lane];
        embsum[b * DEMB + lane] = s;
    }
}

// ---------- k_row: weight_T, denom1, s0, f16 conversions, x transpose ------
__global__ void k_row(const float* __restrict__ x, const float* __restrict__ temb,
                      const float* __restrict__ tt, const float* __restrict__ embsum,
                      _Float16* __restrict__ tembh, _Float16* __restrict__ xT,
                      float* __restrict__ wT, float* __restrict__ denom,
                      float* __restrict__ s0) {
    int idx = blockIdx.x * blockDim.x + threadIdx.x;     // (b,l) flat
    if (idx >= B_ * L_) return;
    int b = idx / L_, l = idx % L_;
    float t = tt[idx];
    const float* xr = x + (size_t)idx * CIN;
    float* wr = wT + (size_t)idx * CIN;
    float s = 0.f;
#pragma unroll
    for (int i = 0; i < 32; i++) {
        float f = expf(-9.210340371976184f * (float)i / 32.0f);  // exp(-ln(1e4)*i/32)
        float a = t * f;
        float sn = sinf(a), cs = cosf(a);
        wr[i] = sn; wr[32 + i] = cs;
        s += xr[i] * sn + xr[32 + i] * cs;
    }
    s0[idx] = s;                                          // s0 = x . weight_T
    const float* er = temb + (size_t)idx * DEMB;
    const float* es = embsum + b * DEMB;
    _Float16* th = tembh + (size_t)idx * DEMB;
    float d = 0.f;
#pragma unroll
    for (int i = 0; i < DEMB; i++) { float v = er[i]; th[i] = (_Float16)v; d += v * es[i]; }
    denom[idx] = d + 1e-8f;                               // distance row-sum + eps
    _Float16* xt = xT + (size_t)b * CIN * L_ + l;
#pragma unroll
    for (int c = 0; c < CIN; c++) xt[(size_t)c * L_] = (_Float16)xr[c];
}

// ---- k_adj: adj_raw = (temb@temb^T)*support (f16), per-row scale ----------
// One block = 16 rows (l0..l0+15) x all 2048 cols; 8 waves x 16 tiles each.
__global__ void k_adj(const _Float16* __restrict__ tembh, const float* __restrict__ sup,
                      const float* __restrict__ denom, _Float16* __restrict__ adj,
                      float* __restrict__ scale) {
    int strip = blockIdx.x;                // 0 .. B*L/16-1
    int b  = strip >> 7;                   // L/16 = 128
    int l0 = (strip & 127) << 4;
    int wave = threadIdx.x >> 5;
    int lane = threadIdx.x & 31;
    int n  = lane & 15;
    int hi = lane >> 4;
    int abase = hi * 8, kbase = hi * 16, moff = hi * 8;

    // A tile: t_emb rows l0..l0+15, K = 0..31 (shared across all m-tiles)
    const _Float16* ar = tembh + ((size_t)b * L_ + l0 + n) * DEMB;
    v8h a0 = *(const v8h*)(ar + kbase);
    v8h a1 = *(const v8h*)(ar + kbase + 8);
    v16h a = __builtin_shufflevector(a0, a1, 0,1,2,3,4,5,6,7,8,9,10,11,12,13,14,15);

    __shared__ float rs[16];
    if (threadIdx.x < 16) rs[threadIdx.x] = 0.f;
    __syncthreads();

    float rowacc[8];
#pragma unroll
    for (int j = 0; j < 8; j++) rowacc[j] = 0.f;

    for (int mt = 0; mt < 16; ++mt) {
        int m0 = (wave * 16 + mt) * 16;
        // B tile: t_emb rows m0..m0+15 transposed (B[k,n] = temb[m0+n][k])
        const _Float16* br = tembh + ((size_t)b * L_ + m0 + n) * DEMB + kbase;
        v16h bb = *(const v16h*)br;
        v8f d = {};
        d = __builtin_amdgcn_wmma_f32_16x16x32_f16(false, a, false, bb, (short)0, d,
                                                   false, false);
        const float*    sr  = sup + ((size_t)b * L_ + l0 + moff) * L_ + m0 + n;
        _Float16*       adr = adj + ((size_t)b * L_ + l0 + moff) * L_ + m0 + n;
#pragma unroll
        for (int j = 0; j < 8; j++) {
            float raw = d[j] * sr[(size_t)j * L_];
            adr[(size_t)j * L_] = (_Float16)raw;
            rowacc[j] += raw;
        }
    }
    // reduce row sums across 16 lanes of each half-wave, accumulate into LDS
#pragma unroll
    for (int j = 0; j < 8; j++) {
        float v = rowacc[j];
        v += __shfl_xor(v, 1); v += __shfl_xor(v, 2);
        v += __shfl_xor(v, 4); v += __shfl_xor(v, 8);
        if (n == 0) atomicAdd(&rs[moff + j], v);
    }
    __syncthreads();
    if (threadIdx.x < 16) {
        int l = l0 + threadIdx.x;
        // adj_norm[l,m] = raw / (rawsum + 1e-8*denom1)   (both eps folded exactly)
        scale[b * L_ + l] = 1.0f / (rs[threadIdx.x] + 1e-8f * denom[b * L_ + l]);
    }
}

// ---- k_spmm: M = adj_raw(16xL) @ Bm(Lx64), row-scaled --------------------
// Block = 8 waves = 2 strips x 4 column-tiles. Each wave owns one 16x16
// output tile and the full K loop; duplicate A-tile reads across the 4 waves
// of a strip are served by WGP$/L2. Per-row dot with weight_T is combined
// across the 4 column waves via an LDS atomic reduction.
template <int SECOND>
__global__ void k_spmm(const _Float16* __restrict__ adj, const _Float16* __restrict__ Bm,
                       const float* __restrict__ wT, const float* __restrict__ scale,
                       const float* __restrict__ s0in, float* __restrict__ sOut,
                       _Float16* __restrict__ T1T) {
    int wave = threadIdx.x >> 5;           // 0..7
    int sloc = wave >> 2;                  // strip within block: 0..1
    int t    = wave & 3;                   // column tile: 0..3
    int strip = blockIdx.x * 2 + sloc;     // 0..B*L/16-1
    int b  = strip >> 7;
    int l0 = (strip & 127) << 4;
    int lane = threadIdx.x & 31;
    int n  = lane & 15;
    int hi = lane >> 4;
    int abase = hi * 8, kbase = hi * 16, moff = hi * 8;

    const _Float16* arow = adj + ((size_t)b * L_ + l0 + n) * L_;
    const _Float16* brow = Bm  + ((size_t)b * CIN + t * 16 + n) * L_;

    __shared__ float sred[2][16];
    if (threadIdx.x < 32) sred[threadIdx.x >> 4][threadIdx.x & 15] = 0.f;
    __syncthreads();

    v8f acc = {};
#pragma unroll 4
    for (int k0 = 0; k0 < L_; k0 += 32) {
        v8h a0 = *(const v8h*)(arow + k0 + kbase);
        v8h a1 = *(const v8h*)(arow + k0 + kbase + 8);
        v16h a = __builtin_shufflevector(a0, a1, 0,1,2,3,4,5,6,7,8,9,10,11,12,13,14,15);
        v16h bb = *(const v16h*)(brow + k0 + kbase);
        acc = __builtin_amdgcn_wmma_f32_16x16x32_f16(false, a, false, bb, (short)0,
                                                     acc, false, false);
    }

    // partial dot with weight_T for this wave's 16 channels (c = 16t + n)
#pragma unroll
    for (int j = 0; j < 8; j++) {
        float v = acc[j] * wT[((size_t)b * L_ + l0 + moff + j) * COUT + t * 16 + n];
        v += __shfl_xor(v, 1); v += __shfl_xor(v, 2);
        v += __shfl_xor(v, 4); v += __shfl_xor(v, 8);
        if (n == 0) atomicAdd(&sred[sloc][moff + j], v);
    }

    if (!SECOND) {
        // store T1^T (col-major f16) for the second SpMM's B operand
        float scl[8];
#pragma unroll
        for (int j = 0; j < 8; j++) scl[j] = scale[b * L_ + l0 + moff + j];
        v8h h;
#pragma unroll
        for (int j = 0; j < 8; j++) h[j] = (_Float16)(acc[j] * scl[j]);
        *(v8h*)(T1T + ((size_t)b * CIN + t * 16 + n) * L_ + l0 + moff) = h;
    }

    __syncthreads();
    if (threadIdx.x < 32) {
        int sloc2  = threadIdx.x >> 4;
        int row    = threadIdx.x & 15;
        int strip2 = blockIdx.x * 2 + sloc2;
        int b2  = strip2 >> 7;
        int l02 = (strip2 & 127) << 4;
        int r = b2 * L_ + l02 + row;
        float v = sred[sloc2][row];
        if (SECOND) sOut[r] = 2.f * scale[r] * v - s0in[r];   // s2
        else        sOut[r] = scale[r] * v;                   // s1
    }
}

// ---- k_out: out = LN(relu(s0*wS0 + s1*wS1 + s2*wS2 + bias) lerp x) --------
__global__ void k_out(const float* __restrict__ x, const int* __restrict__ vidx,
                      const float* __restrict__ wS, const float* __restrict__ bias,
                      const float* __restrict__ lambda_, const float* __restrict__ gamma,
                      const float* __restrict__ beta, const float* __restrict__ s0,
                      const float* __restrict__ s1, const float* __restrict__ s2,
                      float* __restrict__ out) {
    int wave = threadIdx.x >> 5;
    int lane = threadIdx.x & 31;
    int r = blockIdx.x * 8 + wave;          // flat (b,l)
    int l = r & (L_ - 1);
    float a0 = s0[r], a1 = s1[r], a2 = s2[r];
    int vi = vidx[l];
    float lam = lambda_[0];
    const float* w0 = wS + (size_t)vi * COUT;
    const float* w1 = w0 + (size_t)NNODE * COUT;
    const float* w2 = w1 + (size_t)NNODE * COUT;
    const float* xr = x + (size_t)r * CIN;
    float o[2];
#pragma unroll
    for (int q = 0; q < 2; q++) {
        int c = lane + 32 * q;
        float v = a0 * w0[c] + a1 * w1[c] + a2 * w2[c] + bias[c];
        v = fmaxf(v, 0.f);
        o[q] = v * lam + xr[c] * (1.f - lam);
    }
    float sum = o[0] + o[1];
    sum += __shfl_xor(sum, 1);  sum += __shfl_xor(sum, 2);
    sum += __shfl_xor(sum, 4);  sum += __shfl_xor(sum, 8);  sum += __shfl_xor(sum, 16);
    float mu = sum * (1.f / 64.f);
    float d0 = o[0] - mu, d1 = o[1] - mu;
    float sq = d0 * d0 + d1 * d1;
    sq += __shfl_xor(sq, 1);  sq += __shfl_xor(sq, 2);
    sq += __shfl_xor(sq, 4);  sq += __shfl_xor(sq, 8);  sq += __shfl_xor(sq, 16);
    float inv = rsqrtf(sq * (1.f / 64.f) + 1e-5f);
    float* orow = out + (size_t)r * COUT;
    orow[lane]      = d0 * inv * gamma[lane]      + beta[lane];
    orow[lane + 32] = d1 * inv * gamma[lane + 32] + beta[lane + 32];
}

// ---------------------------------------------------------------------------
extern "C" void kernel_launch(void* const* d_in, const int* in_sizes, int n_in,
                              void* d_out, int out_size, void* d_ws, size_t ws_size,
                              hipStream_t stream) {
    const float* x        = (const float*)d_in[0];
    const float* support  = (const float*)d_in[1];
    const float* t_emb    = (const float*)d_in[2];
    const float* tt       = (const float*)d_in[3];
    const int*   vars_idx = (const int*)  d_in[4];
    const float* wS       = (const float*)d_in[5];
    const float* bias     = (const float*)d_in[6];
    const float* lambda_  = (const float*)d_in[7];
    const float* ln_gamma = (const float*)d_in[8];
    const float* ln_beta  = (const float*)d_in[9];
    float* out = (float*)d_out;

    char* ws = (char*)d_ws;
    float*    embsum = (float*)   (ws + OFF_EMBSUM);
    float*    denom  = (float*)   (ws + OFF_DENOM);
    float*    scale  = (float*)   (ws + OFF_SCALE);
    float*    s0p    = (float*)   (ws + OFF_S0);
    float*    s1p    = (float*)   (ws + OFF_S1);
    float*    s2p    = (float*)   (ws + OFF_S2);
    _Float16* tembh  = (_Float16*)(ws + OFF_TEMBH);
    _Float16* xT     = (_Float16*)(ws + OFF_XT);
    float*    wTp    = (float*)   (ws + OFF_WT);
    _Float16* T1Tp   = (_Float16*)(ws + OFF_T1T);
    _Float16* adj    = (_Float16*)(ws + OFF_ADJ);

    k_sum<<<B_, 256, 0, stream>>>(t_emb, embsum);
    k_row<<<(B_ * L_ + 255) / 256, 256, 0, stream>>>(x, t_emb, tt, embsum, tembh, xT,
                                                     wTp, denom, s0p);
    k_adj<<<B_ * L_ / 16, 256, 0, stream>>>(tembh, support, denom, adj, scale);
    // 2 strips per block (8 waves: 2 strips x 4 column tiles)
    k_spmm<0><<<B_ * L_ / 32, 256, 0, stream>>>(adj, xT, wTp, scale, s0p, s1p, T1Tp);
    k_spmm<1><<<B_ * L_ / 32, 256, 0, stream>>>(adj, T1Tp, wTp, scale, s0p, s2p, nullptr);
    k_out<<<B_ * L_ / 8, 256, 0, stream>>>(x, vars_idx, wS, bias, lambda_, ln_gamma,
                                           ln_beta, s0p, s1p, s2p, out);
}
